// MultiLayerRNN_41412074668071
// MI455X (gfx1250) — matrix operand order repaired and checked by reading, not compiled, not run on hardware
//
#include <hip/hip_runtime.h>
#include <hip/hip_bf16.h>

#define B_   128
#define T_   1024
#define DIN_ 256
#define H_   512
#define C_   1000
#define CPAD 1008
#define HS   520   // LDS row stride (bf16 elems): 520*2/4 = 260 ≡ 4 (mod 64) -> conflict-free b128 reads
#define KB0  (DIN_ / 32)   // 8  K-blocks for Wx0
#define KBH  (H_ / 32)     // 16 K-blocks for H-sized weights

typedef __attribute__((ext_vector_type(16))) __bf16 v16bf;
typedef __attribute__((ext_vector_type(8)))  float  v8f;
typedef __attribute__((ext_vector_type(2)))  float  v2f;
typedef unsigned int   u32;
typedef unsigned short u16;
typedef __attribute__((ext_vector_type(4))) u32 v4u;

union AF { v16bf v; v4u q[2]; u32 u[8]; };

__device__ __forceinline__ u16 f2bf(float f) {
  union { float f; u32 u; } x; x.f = f;
  u32 r = x.u + 0x7FFFu + ((x.u >> 16) & 1u);   // round-to-nearest-even
  return (u16)(r >> 16);
}
__device__ __forceinline__ u32 packbf(float lo, float hi) {
  return (u32)f2bf(lo) | ((u32)f2bf(hi) << 16);
}
__device__ __forceinline__ float fast_tanh(float x) {
  float e = __expf(2.0f * x);                   // saturates correctly at +-inf
  return 1.0f - 2.0f / (e + 1.0f);
}
__device__ __forceinline__ v8f wmma_bf16(v16bf a, v16bf b, v8f c) {
  return __builtin_amdgcn_wmma_f32_16x16x32_bf16(false, a, false, b, (short)0, c,
                                                 false, false);
}

// B fragment from fragment-major packed weights: fragment fi = nj*KB + kbi is
// 1 KB contiguous; lane owns 32 contiguous bytes -> two coalesced b128 loads.
__device__ __forceinline__ v16bf load_bp(const u16* __restrict__ Wp, int fi, int lane) {
  AF f;
  const v4u* p = (const v4u*)(Wp + (((size_t)fi << 5) + lane) * 16);
  f.q[0] = p[0]; f.q[1] = p[1];
  return f.v;
}
// A fragment from bf16 matrix (LDS h-state or global), row-major stride ld.
// lane&15 = M row; VGPR0-7 = K+klo..+15, klo=(lane>>4)*16.
__device__ __forceinline__ v16bf load_a(const u16* __restrict__ Hm, int row, int ld,
                                        int kb, int lane) {
  AF f;
  const v4u* p = (const v4u*)(Hm + (size_t)row * ld + kb + ((lane >> 4) << 4));
  f.q[0] = p[0];          // K +0..7
  f.q[1] = p[1];          // K +8..15
  return f.v;
}

// ---------------- prep kernels ----------------
__global__ void k_cvt_pair(const float* __restrict__ src, u32* __restrict__ dst,
                           int npairs) {
  int stride = gridDim.x * blockDim.x;
  const v2f* s = (const v2f*)src;
  for (int i = blockIdx.x * blockDim.x + threadIdx.x; i < npairs; i += stride) {
    v2f p = s[i];
    dst[i] = packbf(p.x, p.y);
  }
}
// Pack W[Nsrc][K] (f32, row-major) into WMMA-B fragment-major bf16:
// dst[((nj*(K/32)+kbi)*32 + lane)*16 + e], n = nj*16+(lane&15),
// k = kbi*32 + (lane>>4)*16 + e.  Rows n >= Nsrc are zero padding.
__global__ void k_packB(const float* __restrict__ W, u16* __restrict__ dst,
                        int Nsrc, int Npad, int K) {
  int total  = Npad * K;
  int stride = gridDim.x * blockDim.x;
  int KBk    = K >> 5;
  for (int i = blockIdx.x * blockDim.x + threadIdx.x; i < total; i += stride) {
    int e    = i & 15;
    int lane = (i >> 4) & 31;
    int fi   = i >> 9;              // nj*KBk + kbi
    int kbi  = fi % KBk;
    int nj   = fi / KBk;
    int n    = nj * 16 + (lane & 15);
    int k    = kbi * 32 + ((lane >> 4) << 4) + e;
    dst[i] = (n < Nsrc) ? f2bf(W[(size_t)n * K + k]) : (u16)0;
  }
}

// ---------------- persistent recurrent kernel ----------------
// 8 blocks (one per 16-row batch tile) x 512 threads (16 waves).
// Wave w computes output-column tiles {w, w+16} (cols n0, n0+256), sharing A.
// h0 (ping-pong) and h1 live entirely in LDS; sync = 3 s_barriers/step.
// Loop shape kept low-pressure so the t-invariant Wx0 B-fragments stay
// hoisted in VGPRs without spilling; next-step x rows are prefetched so
// layer-0's A loads hit WGP$/L2.
__launch_bounds__(512, 1)
__global__ void k_rnn(const u16* __restrict__ xbf,
                      const u16* __restrict__ Wx0, const u16* __restrict__ Wh0,
                      const u16* __restrict__ Wx1, const u16* __restrict__ Wh1,
                      const float* __restrict__ bx0, const float* __restrict__ bh0,
                      const float* __restrict__ bx1, const float* __restrict__ bh1,
                      u16* __restrict__ h1g) {
  __shared__ u16 h0s[2][16 * HS];
  __shared__ u16 h1s[16 * HS];

  const int tid  = threadIdx.x;
  const int lane = tid & 31;
  const int wave = tid >> 5;          // 0..15
  const int mi   = blockIdx.x;        // batch tile 0..7
  const int nl   = lane & 15;         // A row (local batch) / B,C col-in-tile
  const int n0   = wave * 16 + nl;    // first owned output column
  const int n1   = n0 + 256;          // second owned output column
  const int m0   = (lane >> 4) << 3;  // C/D row base for this lane
  const int nt0  = wave;              // first owned column tile
  const int nt1  = wave + 16;         // second owned column tile

  const float b0a = bx0[n0] + bh0[n0], b0b = bx0[n1] + bh0[n1];
  const float b1a = bx1[n0] + bh1[n0], b1b = bx1[n1] + bh1[n1];

  for (int i = tid; i < 16 * HS; i += 512) { h0s[0][i] = 0; h1s[i] = 0; }
  __syncthreads();

  const int  xrow  = mi * 16 + nl;
  const u16* xbase = xbf + (size_t)xrow * T_ * DIN_;
  const int  pfo   = (lane >> 4) << 6;   // lane-pair covers the 512B row in 128B chunks

  for (int t = 0; t < T_; ++t) {
    const u16* h0c = h0s[t & 1];
    u16*       h0n = h0s[(t + 1) & 1];

    // ---- layer 0: tanh(x_t Wx0^T + h0 Wh0^T + b) ----
    v8f acc0, acc1;
#pragma unroll
    for (int r = 0; r < 8; ++r) { acc0[r] = b0a; acc1[r] = b0b; }

    // x part: Wx0 B fragments are t-invariant -> compiler keeps them in VGPRs.
    {
      const u16* xp = xbase + (size_t)t * DIN_;
#pragma unroll
      for (int i2 = 0; i2 < KB0; i2 += 4) {
        AF xa[4];
#pragma unroll
        for (int j = 0; j < 4; ++j)
          xa[j].v = load_a(xp, 0, DIN_, (i2 + j) * 32, lane);
#pragma unroll
        for (int j = 0; j < 4; ++j) {
          acc0 = wmma_bf16(xa[j].v, load_bp(Wx0, nt0 * KB0 + i2 + j, lane), acc0);
          acc1 = wmma_bf16(xa[j].v, load_bp(Wx0, nt1 * KB0 + i2 + j, lane), acc1);
        }
      }
      // warm next step's x row while h0/h1 compute + barriers run
      const u16* xn = xbase + (size_t)((t + 1 < T_) ? t + 1 : t) * DIN_ + pfo;
      __builtin_prefetch(xn, 0, 3);
      __builtin_prefetch(xn + 128, 0, 3);
    }
    // h0 part: A from LDS, B streamed coalesced from L2 (packed fragments).
#pragma unroll 4
    for (int kbi = 0; kbi < KBH; ++kbi) {
      v16bf a = load_a(h0c, nl, HS, kbi * 32, lane);
      acc0 = wmma_bf16(a, load_bp(Wh0, nt0 * KBH + kbi, lane), acc0);
      acc1 = wmma_bf16(a, load_bp(Wh0, nt1 * KBH + kbi, lane), acc1);
    }
#pragma unroll
    for (int r = 0; r < 8; ++r) {
      h0n[(m0 + r) * HS + n0] = f2bf(fast_tanh(acc0[r]));
      h0n[(m0 + r) * HS + n1] = f2bf(fast_tanh(acc1[r]));
    }
    __syncthreads();                       // new h0 visible

    // ---- layer 1: tanh(h0 Wx1^T + h1 Wh1^T + b) ----
#pragma unroll
    for (int r = 0; r < 8; ++r) { acc0[r] = b1a; acc1[r] = b1b; }
#pragma unroll 4
    for (int kbi = 0; kbi < KBH; ++kbi) {
      v16bf a = load_a(h0n, nl, HS, kbi * 32, lane);
      acc0 = wmma_bf16(a, load_bp(Wx1, nt0 * KBH + kbi, lane), acc0);
      acc1 = wmma_bf16(a, load_bp(Wx1, nt1 * KBH + kbi, lane), acc1);
    }
#pragma unroll 4
    for (int kbi = 0; kbi < KBH; ++kbi) {
      v16bf a = load_a(h1s, nl, HS, kbi * 32, lane);
      acc0 = wmma_bf16(a, load_bp(Wh1, nt0 * KBH + kbi, lane), acc0);
      acc1 = wmma_bf16(a, load_bp(Wh1, nt1 * KBH + kbi, lane), acc1);
    }
    __syncthreads();                       // all reads of old h1 done
#pragma unroll
    for (int r = 0; r < 8; ++r) {
      h1s[(m0 + r) * HS + n0] = f2bf(fast_tanh(acc0[r]));
      h1s[(m0 + r) * HS + n1] = f2bf(fast_tanh(acc1[r]));
    }
    __syncthreads();                       // new h1 visible
  }

  // export final h1 (dense [B][H] bf16) for the FC kernel
  for (int i = tid; i < 16 * H_; i += 512) {
    int m = i >> 9, c = i & (H_ - 1);
    h1g[(size_t)(mi * 16 + m) * H_ + c] = h1s[m * HS + c];
  }
}

// ---------------- final projection: out = h1 fc^T + b ----------------
__launch_bounds__(256)
__global__ void k_fc(const u16* __restrict__ h1g, const u16* __restrict__ FCp,
                     const float* __restrict__ fcb, float* __restrict__ out) {
  const int tid = threadIdx.x, lane = tid & 31, mi = tid >> 5;  // 8 waves
  const int cj  = blockIdx.x;                                   // 0..62
  const int nl  = lane & 15;
  const int n   = cj * 16 + nl;                                 // 0..1007
  const int m0  = (lane >> 4) << 3;
  v8f acc;
  float bv = (n < C_) ? fcb[n] : 0.0f;
#pragma unroll
  for (int r = 0; r < 8; ++r) acc[r] = bv;
  const int row = mi * 16 + nl;
#pragma unroll 4
  for (int kbi = 0; kbi < KBH; ++kbi)
    acc = wmma_bf16(load_a(h1g, row, H_, kbi * 32, lane),
                    load_bp(FCp, cj * KBH + kbi, lane), acc);
  if (n < C_) {
#pragma unroll
    for (int r = 0; r < 8; ++r)
      out[(size_t)(mi * 16 + m0 + r) * C_ + n] = acc[r];
  }
}

extern "C" void kernel_launch(void* const* d_in, const int* in_sizes, int n_in,
                              void* d_out, int out_size, void* d_ws, size_t ws_size,
                              hipStream_t stream) {
  (void)in_sizes; (void)n_in; (void)out_size; (void)ws_size;
  const float* x    = (const float*)d_in[0];
  const float* Wx0w = (const float*)d_in[1];
  const float* Wx0b = (const float*)d_in[2];
  const float* Wh0w = (const float*)d_in[3];
  const float* Wh0b = (const float*)d_in[4];
  const float* Wx1w = (const float*)d_in[5];
  const float* Wx1b = (const float*)d_in[6];
  const float* Wh1w = (const float*)d_in[7];
  const float* Wh1b = (const float*)d_in[8];
  const float* fcw  = (const float*)d_in[9];
  const float* fcb  = (const float*)d_in[10];

  // workspace layout (bytes), total ~70.1 MB
  char* ws  = (char*)d_ws;
  u16* xbf  = (u16*)(ws + 0);              // 128*1024*256*2 = 67108864
  u16* pWx0 = (u16*)(ws + 67108864);       // 512*256*2 = 262144
  u16* pWh0 = (u16*)(ws + 67371008);       // 512*512*2 = 524288
  u16* pWx1 = (u16*)(ws + 67895296);
  u16* pWh1 = (u16*)(ws + 68419584);
  u16* pFC  = (u16*)(ws + 68943872);       // 1008*512*2 = 1032192
  u16* h1g  = (u16*)(ws + 69976064);       // 128*512*2  = 131072

  // one-time parallel conversions / fragment packing (off the critical path)
  k_cvt_pair<<<2048, 256, 0, stream>>>(x, (u32*)xbf, B_ * T_ * DIN_ / 2);
  k_packB<<<256, 256, 0, stream>>>(Wx0w, pWx0, H_, H_,   DIN_);
  k_packB<<<256, 256, 0, stream>>>(Wh0w, pWh0, H_, H_,   H_);
  k_packB<<<256, 256, 0, stream>>>(Wx1w, pWx1, H_, H_,   H_);
  k_packB<<<256, 256, 0, stream>>>(Wh1w, pWh1, H_, H_,   H_);
  k_packB<<<256, 256, 0, stream>>>(fcw,  pFC,  C_, CPAD, H_);

  k_rnn<<<8, 512, 0, stream>>>(xbf, pWx0, pWh0, pWx1, pWh1,
                               Wx0b, Wh0b, Wx1b, Wh1b, h1g);
  k_fc <<<63, 256, 0, stream>>>(h1g, pFC, fcb, (float*)d_out);
}
